// GptOssExperts_19954418057881
// MI455X (gfx1250) — matrix operand order repaired and checked by reading, not compiled, not run on hardware
//
#include <hip/hip_runtime.h>
#include <hip/hip_bf16.h>
#include <math.h>

// GPT-OSS MoE experts (dense routing over 8 experts), MI455X / gfx1250.
// fp32 -> bf16 convert/transpose pass, then two bf16 WMMA GEMMs
// (v_wmma_f32_16x16x32_bf16, wave32). B tiles are staged into LDS with
// GLOBAL_LOAD_ASYNC_TO_LDS_B128 (ASYNCcnt), double-buffered so the DMA of
// tile t+1 overlaps the WMMA work on tile t.
//
// Workspace layout (bytes, needs ~451 MB):
//   xb   : 1024*2880*2            =   5,898,240
//   wgt  : 8*5760*2880*2          = 265,420,800   (transposed + de-interleaved)
//   wdt  : 8*2880*2880*2          = 132,710,400   (transposed)
//   act  : 8*1024*2880*2          =  47,185,920   (rw-scaled activations)

#define ALPHA 1.702f
#define LIMIT 7.0f
#define HDIM  2880
#define DDIM  2880
#define NEXP  8
#define NTOK  1024
#define NGU   5760  // 2*DDIM

typedef __attribute__((ext_vector_type(16))) __bf16 bf16x16;
typedef __attribute__((ext_vector_type(8)))  __bf16 bf16x8;
typedef __attribute__((ext_vector_type(8)))  float  f32x8;

union BFrag { bf16x16 v; bf16x8 h[2]; };

// A fragment (16x32 bf16, row-major source, row pointer already at k):
// lane L (row = L%16): elems i = K 16*(L/16)+i
__device__ __forceinline__ bf16x16 load_a16x32(const __bf16* rowk, int hsel) {
  BFrag f;
  f.h[0] = *reinterpret_cast<const bf16x8*>(rowk + 16 * hsel);
  f.h[1] = *reinterpret_cast<const bf16x8*>(rowk + 16 * hsel + 8);
  return f.v;
}

// B fragment (32x16 bf16) from a K-contiguous column: lane L (col = L%16):
// elems i = K 16*(L/16)+i -> 32 contiguous bytes (two b128 loads).
__device__ __forceinline__ bf16x16 load_b32x16(const __bf16* colk) {
  BFrag f;
  f.h[0] = *reinterpret_cast<const bf16x8*>(colk);
  f.h[1] = *reinterpret_cast<const bf16x8*>(colk + 8);
  return f.v;
}

__device__ __forceinline__ f32x8 wmma_bf16(bf16x16 a, bf16x16 b, f32x8 c) {
  return __builtin_amdgcn_wmma_f32_16x16x32_bf16(
      /*neg_a=*/false, a, /*neg_b=*/false, b,
      /*c_mod=*/(short)0, c, /*reuse_a=*/false, /*reuse_b=*/false);
}

// ---- gfx1250 async global->LDS copy (per lane 16B), tracked by ASYNCcnt ----
// Flat pointers to LDS keep the LDS byte offset in addr[31:0] (ISA aperture
// mapping), so the low 32 bits are the VDST operand value.
__device__ __forceinline__ unsigned lds_addr32(const void* p) {
  return (unsigned)(size_t)p;
}
__device__ __forceinline__ void async_copy_b128(unsigned lds_off, const void* g) {
  asm volatile("global_load_async_to_lds_b128 %0, %1, off"
               :: "v"(lds_off), "v"((unsigned long long)(size_t)g)
               : "memory");
}
__device__ __forceinline__ void wait_async0() {
  asm volatile("s_wait_asynccnt 0x0" ::: "memory");
}

// ---------------------------------------------------------------- conversions

__global__ void cvt_f32_bf16(const float* __restrict__ in,
                             __bf16* __restrict__ out, int n) {
  int i = blockIdx.x * blockDim.x + threadIdx.x;
  if (i < n) out[i] = (__bf16)in[i];
}

// in: [e][R][C] fp32  ->  out: [e][C'][R] bf16 (transposed, optional
// gate/up de-interleave on the C index). 32x32 tiles via LDS.
template <bool DEINT>
__global__ void transpose_cvt(const float* __restrict__ in,
                              __bf16* __restrict__ out, int R, int C) {
  __shared__ float tile[32][33];
  const int ex = blockIdx.z;
  const int c0 = blockIdx.x * 32, r0 = blockIdx.y * 32;
  const int tx = threadIdx.x, ty = threadIdx.y;
  const float* ip = in + (size_t)ex * R * C;
  __bf16* op = out + (size_t)ex * R * C;
#pragma unroll
  for (int i = 0; i < 4; i++)
    tile[ty + 8 * i][tx] = ip[(size_t)(r0 + ty + 8 * i) * C + c0 + tx];
  __syncthreads();
#pragma unroll
  for (int i = 0; i < 4; i++) {
    const int c = c0 + ty + 8 * i;
    const int cp = DEINT ? ((c >> 1) + (c & 1) * (C >> 1)) : c;
    op[(size_t)cp * R + r0 + tx] = (__bf16)tile[tx][ty + 8 * i];
  }
}

// ------------------------------------------------------- GEMM1: x @ Wg + GLU
// Workgroup: 8 waves as 4(M) x 2(N); wave tile = 64 rows x (32 gate + 32 up).
// B tile (64 gate + 64 up cols x 32 K) async-staged into LDS, double-buffered.
__global__ __launch_bounds__(256) void gemm1_gateup(
    const __bf16* __restrict__ xb,   // [1024][2880]
    const __bf16* __restrict__ wgt,  // [e][n'][2880], n'<2880 gate, else up
    const float* __restrict__ gub,   // [e][5760] interleaved bias
    const float* __restrict__ rw,    // [1024][8]
    __bf16* __restrict__ act)        // [e][1024][2880]
{
  // [buf][col][k] bf16: col 0..63 gate, 64..127 up; 32 K per col (64B).
  __shared__ __align__(16) __bf16 bsh[2][128 * 32];

  const int tid  = threadIdx.x;
  const int e    = blockIdx.z;
  const int wave = tid >> 5;
  const int lane = tid & 31;
  const int lr   = lane & 15;
  const int hsel = lane >> 4;
  const int wm = wave >> 1, wn = wave & 1;
  const int t0  = blockIdx.x * 256 + wm * 64;
  const int dwg = blockIdx.y * 64;  // workgroup act-col base

  const __bf16* wg_base = wgt + (size_t)e * NGU * HDIM;

  // async-copy role: thread -> (column, 32B half-row); 8KB tile / 256 thr.
  const int ccol = tid >> 1;                 // 0..127
  const int choff = (tid & 1) * 16;          // halves: 0 or 16
  const __bf16* gsrc_col =
      (ccol < 64) ? (wg_base + (size_t)(dwg + ccol) * HDIM)
                  : (wg_base + (size_t)(DDIM + dwg + (ccol - 64)) * HDIM);

  f32x8 cg[4][2], cu[4][2];
#pragma unroll
  for (int i = 0; i < 4; i++)
#pragma unroll
    for (int j = 0; j < 2; j++) { cg[i][j] = (f32x8)0.f; cu[i][j] = (f32x8)0.f; }

  const __bf16* xrow[4];
#pragma unroll
  for (int tm = 0; tm < 4; tm++)
    xrow[tm] = xb + (size_t)(t0 + tm * 16 + lr) * HDIM;

  // prologue: stage tile 0
  {
    const __bf16* src = gsrc_col + choff;
    unsigned dst = lds_addr32(&bsh[0][ccol * 32 + choff]);
    async_copy_b128(dst, src);
    async_copy_b128(dst + 16, src + 8);
  }

#pragma unroll 1
  for (int kt = 0; kt < HDIM / 32; ++kt) {
    wait_async0();     // this wave's pieces of tile kt landed
    __syncthreads();   // all waves' pieces visible; prev reads of other buf done
    if (kt + 1 < HDIM / 32) {
      const __bf16* src = gsrc_col + (kt + 1) * 32 + choff;
      unsigned dst = lds_addr32(&bsh[(kt + 1) & 1][ccol * 32 + choff]);
      async_copy_b128(dst, src);
      async_copy_b128(dst + 16, src + 8);
    }
    const int k = kt * 32;
    bf16x16 a[4];
#pragma unroll
    for (int tm = 0; tm < 4; tm++) a[tm] = load_a16x32(xrow[tm] + k, hsel);
    const __bf16* bl = &bsh[kt & 1][0];
#pragma unroll
    for (int tn = 0; tn < 2; tn++) {
      const int cl = wn * 32 + tn * 16 + lr;  // col-local 0..63
      bf16x16 bg = load_b32x16(bl + cl * 32 + 16 * hsel);
      bf16x16 bu = load_b32x16(bl + (64 + cl) * 32 + 16 * hsel);
#pragma unroll
      for (int tm = 0; tm < 4; tm++) {
        cg[tm][tn] = wmma_bf16(a[tm], bg, cg[tm][tn]);
        cu[tm][tn] = wmma_bf16(a[tm], bu, cu[tm][tn]);
      }
    }
  }

#pragma unroll
  for (int tn = 0; tn < 2; tn++) {
    const int d = dwg + wn * 32 + tn * 16 + lr;
    const float bgb = gub[(size_t)e * NGU + 2 * d];
    const float bub = gub[(size_t)e * NGU + 2 * d + 1];
#pragma unroll
    for (int tm = 0; tm < 4; tm++) {
#pragma unroll
      for (int v = 0; v < 8; v++) {
        const int t = t0 + tm * 16 + v + 8 * hsel;
        float g = cg[tm][tn][v] + bgb;
        float u = cu[tm][tn][v] + bub;
        g = fminf(g, LIMIT);
        u = fminf(fmaxf(u, -LIMIT), LIMIT);
        const float glu = g * (1.f / (1.f + __expf(-ALPHA * g)));
        const float av = (u + 1.f) * glu * rw[t * NEXP + e];
        act[((size_t)e * NTOK + t) * DDIM + d] = (__bf16)av;
      }
    }
  }
}

// ------------------------------------------- GEMM2: sum_e act'_e @ Wd_e + b
// Single GEMM with K = 8*2880 (rw folded into act in GEMM1).
// Workgroup: 8 waves as 8(M) x 1(N); wave tile = 64x64.
// B tile (64 cols x 32 K) async-staged into LDS, double-buffered.
__global__ __launch_bounds__(256) void gemm2_down(
    const __bf16* __restrict__ act,   // [e][1024][2880]
    const __bf16* __restrict__ wdt,   // [e][h][2880]
    const float* __restrict__ rw,     // [1024][8]
    const float* __restrict__ dbias,  // [e][2880]
    float* __restrict__ out)          // [1024][2880]
{
  __shared__ __align__(16) __bf16 bsh[2][64 * 32];  // [buf][col][k]

  const int tid  = threadIdx.x;
  const int wave = tid >> 5;
  const int lane = tid & 31;
  const int lr = lane & 15, hsel = lane >> 4;
  const int t0 = blockIdx.x * 512 + wave * 64;
  const int h0 = blockIdx.y * 64;

  // async-copy role: thread -> (column, 16B quarter-row); 4KB tile / 256 thr.
  const int ccol = tid >> 2;            // 0..63
  const int choff = (tid & 3) * 8;      // halves: 0,8,16,24
  const int NT = NEXP * (DDIM / 32);    // 720 tiles

  f32x8 c[4][4];
#pragma unroll
  for (int i = 0; i < 4; i++)
#pragma unroll
    for (int j = 0; j < 4; j++) c[i][j] = (f32x8)0.f;

  // prologue: stage tile 0 (e=0, k=0)
  {
    const __bf16* src = wdt + (size_t)(h0 + ccol) * DDIM + choff;
    async_copy_b128(lds_addr32(&bsh[0][ccol * 32 + choff]), src);
  }

#pragma unroll 1
  for (int tt = 0; tt < NT; ++tt) {
    wait_async0();
    __syncthreads();
    if (tt + 1 < NT) {
      const int en = (tt + 1) / (DDIM / 32);
      const int kn = ((tt + 1) - en * (DDIM / 32)) * 32;
      const __bf16* src =
          wdt + ((size_t)en * HDIM + h0 + ccol) * DDIM + kn + choff;
      async_copy_b128(lds_addr32(&bsh[(tt + 1) & 1][ccol * 32 + choff]), src);
    }
    const int e = tt / (DDIM / 32);
    const int k = (tt - e * (DDIM / 32)) * 32;
    const __bf16* abase = act + (size_t)e * NTOK * DDIM;
    bf16x16 a[4];
#pragma unroll
    for (int tm = 0; tm < 4; tm++)
      a[tm] = load_a16x32(abase + (size_t)(t0 + tm * 16 + lr) * DDIM + k, hsel);
    const __bf16* bl = &bsh[tt & 1][0];
#pragma unroll
    for (int tn = 0; tn < 4; tn++) {
      bf16x16 b = load_b32x16(bl + (tn * 16 + lr) * 32 + 16 * hsel);
#pragma unroll
      for (int tm = 0; tm < 4; tm++) c[tm][tn] = wmma_bf16(a[tm], b, c[tm][tn]);
    }
  }

#pragma unroll
  for (int tn = 0; tn < 4; tn++) {
    const int h = h0 + tn * 16 + lr;
    float db[NEXP];
#pragma unroll
    for (int e = 0; e < NEXP; e++) db[e] = dbias[(size_t)e * HDIM + h];
#pragma unroll
    for (int tm = 0; tm < 4; tm++) {
#pragma unroll
      for (int v = 0; v < 8; v++) {
        const int t = t0 + tm * 16 + v + 8 * hsel;
        float acc = c[tm][tn][v];
#pragma unroll
        for (int e = 0; e < NEXP; e++) acc += rw[t * NEXP + e] * db[e];
        out[(size_t)t * HDIM + h] = acc;
      }
    }
  }
}

// ------------------------------------------------------------------- launch

extern "C" void kernel_launch(void* const* d_in, const int* in_sizes, int n_in,
                              void* d_out, int out_size, void* d_ws,
                              size_t ws_size, hipStream_t stream) {
  (void)in_sizes; (void)n_in; (void)out_size; (void)ws_size;
  const float* hs    = (const float*)d_in[0];  // (1,1024,2880)
  const float* rw    = (const float*)d_in[1];  // (1024,8)
  const float* gup   = (const float*)d_in[2];  // (8,2880,5760)
  const float* gub   = (const float*)d_in[3];  // (8,5760)
  const float* dwp   = (const float*)d_in[4];  // (8,2880,2880)
  const float* dbias = (const float*)d_in[5];  // (8,2880)
  float* out = (float*)d_out;

  __bf16* xb  = (__bf16*)d_ws;
  __bf16* wgt = xb + (size_t)NTOK * HDIM;
  __bf16* wdt = wgt + (size_t)NEXP * NGU * HDIM;
  __bf16* act = wdt + (size_t)NEXP * HDIM * DDIM;

  const int nx = NTOK * HDIM;
  cvt_f32_bf16<<<(nx + 255) / 256, 256, 0, stream>>>(hs, xb, nx);
  transpose_cvt<true><<<dim3(NGU / 32, HDIM / 32, NEXP), dim3(32, 8), 0, stream>>>(
      gup, wgt, HDIM, NGU);
  transpose_cvt<false><<<dim3(HDIM / 32, DDIM / 32, NEXP), dim3(32, 8), 0, stream>>>(
      dwp, wdt, DDIM, HDIM);

  gemm1_gateup<<<dim3(NTOK / 256, DDIM / 64, NEXP), 256, 0, stream>>>(
      xb, wgt, gub, rw, act);
  gemm2_down<<<dim3(NTOK / 512, HDIM / 64), 256, 0, stream>>>(
      act, wdt, rw, dbias, out);
}
